// MLA_45148696217039
// MI455X (gfx1250) — matrix operand order repaired and checked by reading, not compiled, not run on hardware
//
#include <hip/hip_runtime.h>
#include <cstdint>
#include <cstddef>

// ---------------------------------------------------------------------------
// MLA forward for MI455X (gfx1250, wave32).  All GEMMs + attention on
// v_wmma_f32_16x16x32_bf16 (bf16 inputs, f32 accumulation).  Tile staging
// uses GLOBAL_LOAD_ASYNC_TO_LDS (ASYNCcnt) when the toolchain exposes it.
// ---------------------------------------------------------------------------

#define DEVFN __device__ __forceinline__

typedef __attribute__((ext_vector_type(16))) __bf16 bf16x16;
typedef __attribute__((ext_vector_type(8)))  float  f32x8;
typedef __attribute__((ext_vector_type(4)))  int    v4i;
typedef __attribute__((ext_vector_type(2)))  int    v2i;

union Frag16 {            // 32 bytes: one WMMA 16-bit A/B fragment per lane
  uint4   q[2];
  bf16x16 v;
};

static constexpr int Bc  = 2;
static constexpr int Tc  = 2048;
static constexpr int Cc  = 2048;
static constexpr int Hc  = 16;
static constexpr int DNc = 128;
static constexpr int DRc = 64;
static constexpr int DVc = 128;
static constexpr int QLc = 1536;
static constexpr int KVLc = 512;
static constexpr int BT  = Bc * Tc;          // 4096 rows
static constexpr int DQK = DNc + DRc;        // 192
static constexpr float LN_EPS = 1e-5f;

// ---- async global->LDS helpers (CDNA5 ASYNCcnt path, with safe fallback) ---
#if defined(__has_builtin)
#  if __has_builtin(__builtin_amdgcn_global_load_async_to_lds_b128)
#    define HAVE_ASYNC_B128 1
#  else
#    define HAVE_ASYNC_B128 0
#  endif
#  if __has_builtin(__builtin_amdgcn_global_load_async_to_lds_b64)
#    define HAVE_ASYNC_B64 1
#  else
#    define HAVE_ASYNC_B64 0
#  endif
#else
#  define HAVE_ASYNC_B128 0
#  define HAVE_ASYNC_B64 0
#endif
#define HAVE_ASYNC_ANY (HAVE_ASYNC_B128 || HAVE_ASYNC_B64)

typedef __attribute__((address_space(1))) v4i* g4p;   // global v4i*
typedef __attribute__((address_space(3))) v4i* l4p;   // LDS v4i*
typedef __attribute__((address_space(1))) v2i* g2p;   // global v2i*
typedef __attribute__((address_space(3))) v2i* l2p;   // LDS v2i*

DEVFN void g2l_b128(const void* g, void* l) {
#if HAVE_ASYNC_B128
  __builtin_amdgcn_global_load_async_to_lds_b128((g4p)(void*)g, (l4p)l, 0, 0);
#else
  *(uint4*)l = *(const uint4*)g;
#endif
}

DEVFN void g2l_b64(const void* g, void* l) {
#if HAVE_ASYNC_B64
  __builtin_amdgcn_global_load_async_to_lds_b64((g2p)(void*)g, (l2p)l, 0, 0);
#else
  *(uint2*)l = *(const uint2*)g;
#endif
}

DEVFN void async_wait() {
#if HAVE_ASYNC_ANY
#  if __has_builtin(__builtin_amdgcn_s_wait_asynccnt)
  __builtin_amdgcn_s_wait_asynccnt(0);
#  else
  asm volatile("s_wait_asynccnt 0" ::: "memory");
#  endif
#endif
}

DEVFN unsigned short f2bf(float f) {
  union { float f; unsigned int u; } c; c.f = f;
  unsigned int u = c.u;
  unsigned int r = u + 0x7FFFu + ((u >> 16) & 1u);   // round-to-nearest-even
  if ((u & 0x7F800000u) == 0x7F800000u) r = u;        // inf/nan passthrough
  return (unsigned short)(r >> 16);
}

// ---------------------------------------------------------------------------
// f32 -> bf16 convert (grid-stride)
// ---------------------------------------------------------------------------
__global__ void cvt_f32_bf16(const float* __restrict__ in,
                             unsigned short* __restrict__ out, size_t n) {
  size_t i = (size_t)blockIdx.x * blockDim.x + threadIdx.x;
  size_t stride = (size_t)gridDim.x * blockDim.x;
  for (; i < n; i += stride) out[i] = f2bf(in[i]);
}

// ---------------------------------------------------------------------------
// Generic bf16 GEMM:  C[M,N] = A[M,K] * B[N,K]^T (+ bias[N])
// Block: 256 threads (8 waves), tile 128(M) x 64(N), K staged 32-deep in LDS.
// Requires: M%128==0, N%64==0, K%32==0  (true for all calls here).
// ---------------------------------------------------------------------------
__global__ __launch_bounds__(256) void gemm_bf16_nt(
    const unsigned short* __restrict__ A,
    const unsigned short* __restrict__ Bm,
    const float* __restrict__ bias,
    float* __restrict__ Cm,
    int M, int N, int K)
{
  __shared__ unsigned short As[128 * 40];   // stride 40 halfs (80B) -> no bank conflicts
  __shared__ unsigned short Bs[64 * 40];

  const int tid  = threadIdx.x;
  const int wave = tid >> 5, lane = tid & 31;
  const int half = lane >> 4, l15 = lane & 15;
  const int wm = wave & 3, wn = wave >> 2;           // 4x2 wave grid
  const int m0 = blockIdx.y * 128, n0 = blockIdx.x * 64;

  const f32x8 fz = {0.f,0.f,0.f,0.f,0.f,0.f,0.f,0.f};
  f32x8 acc[2][2] = {{fz, fz}, {fz, fz}};

  const int ar = tid >> 1, ac = (tid & 1) * 16;      // 128 rows x 2 chunks of 16
  const int br = tid >> 2, bc = (tid & 3) * 8;       // 64 rows x 4 chunks of 8
  const unsigned short* Ag = A  + (size_t)(m0 + ar) * K + ac;
  const unsigned short* Bg = Bm + (size_t)(n0 + br) * K + bc;
  unsigned short* Al = &As[ar * 40 + ac];
  unsigned short* Bl = &Bs[br * 40 + bc];

  for (int k0 = 0; k0 < K; k0 += 32) {
    g2l_b128(Ag + k0, Al);
    g2l_b64 (Bg + k0, Bl);
    if (k0 + 32 < K) {
      __builtin_prefetch(Ag + k0 + 32, 0, 3);
      __builtin_prefetch(Bg + k0 + 32, 0, 3);
    }
    async_wait();
    __syncthreads();

    Frag16 af[2], bf[2];
#pragma unroll
    for (int mi = 0; mi < 2; ++mi) {      // A 16x32: K = half*8..+7 and +16..+23
      const unsigned short* p = &As[(wm * 32 + mi * 16 + l15) * 40 + half * 8];
      af[mi].q[0] = *(const uint4*)p;
      af[mi].q[1] = *(const uint4*)(p + 16);
    }
#pragma unroll
    for (int ni = 0; ni < 2; ++ni) {      // B 32x16: K = half*8..+7 and +16..+23
      const unsigned short* p = &Bs[(wn * 32 + ni * 16 + l15) * 40 + half * 8];
      bf[ni].q[0] = *(const uint4*)p;
      bf[ni].q[1] = *(const uint4*)(p + 16);
    }
#pragma unroll
    for (int mi = 0; mi < 2; ++mi)
#pragma unroll
      for (int ni = 0; ni < 2; ++ni)
        acc[mi][ni] = __builtin_amdgcn_wmma_f32_16x16x32_bf16(
            false, af[mi].v, false, bf[ni].v, (short)0, acc[mi][ni], false, false);
    __syncthreads();
  }

#pragma unroll
  for (int mi = 0; mi < 2; ++mi)
#pragma unroll
    for (int ni = 0; ni < 2; ++ni)
#pragma unroll
      for (int r = 0; r < 8; ++r) {
        int row = m0 + wm * 32 + mi * 16 + r + half * 8;
        int col = n0 + wn * 32 + ni * 16 + l15;
        float v = acc[mi][ni][r];
        if (bias) v += bias[col];
        Cm[(size_t)row * N + col] = v;
      }
}

// ---------------------------------------------------------------------------
// Row LayerNorm: block per row; optional bf16 and/or f32 outputs.
// ---------------------------------------------------------------------------
__global__ __launch_bounds__(256) void ln_kernel(
    const float* __restrict__ in, int in_stride, int cols,
    const float* __restrict__ g, const float* __restrict__ b,
    unsigned short* __restrict__ out_bf, int bf_stride,
    float* __restrict__ out_f32, int f32_stride)
{
  __shared__ float red[256];
  const int row = blockIdx.x, tid = threadIdx.x;
  const float* x = in + (size_t)row * in_stride;

  float s = 0.f;
  for (int c = tid; c < cols; c += 256) s += x[c];
  red[tid] = s; __syncthreads();
  for (int st = 128; st > 0; st >>= 1) { if (tid < st) red[tid] += red[tid + st]; __syncthreads(); }
  float mu = red[0] / (float)cols; __syncthreads();

  float v = 0.f;
  for (int c = tid; c < cols; c += 256) { float d = x[c] - mu; v += d * d; }
  red[tid] = v; __syncthreads();
  for (int st = 128; st > 0; st >>= 1) { if (tid < st) red[tid] += red[tid + st]; __syncthreads(); }
  float rstd = rsqrtf(red[0] / (float)cols + LN_EPS);

  for (int c = tid; c < cols; c += 256) {
    float y = (x[c] - mu) * rstd * g[c] + b[c];
    if (out_bf)  out_bf [(size_t)row * bf_stride  + c] = f2bf(y);
    if (out_f32) out_f32[(size_t)row * f32_stride + c] = y;
  }
}

// ---------------------------------------------------------------------------
// q repack: q_raw (BT x H*192 f32) -> Qf bf16 [(b,h,t) x 192], rope on last 64
// ---------------------------------------------------------------------------
__global__ __launch_bounds__(256) void q_repack(
    const float* __restrict__ q_raw, unsigned short* __restrict__ Qf)
{
  const int row = blockIdx.x, b = row / Tc, t = row % Tc, tid = threadIdx.x;
  const float* src = q_raw + (size_t)row * (Hc * DQK);
  for (int i = tid; i < Hc * DNc; i += 256) {
    int h = i >> 7, c = i & 127;
    Qf[(((size_t)(b * Hc + h)) * Tc + t) * DQK + c] = f2bf(src[h * DQK + c]);
  }
  for (int p = tid; p < Hc * (DRc / 2); p += 256) {
    int h = p >> 5, i = p & 31;
    float freq = __expf(-__logf(10000.0f) * (float)(2 * i) / (float)DRc);
    float ang = (float)t * freq;
    float cs = __cosf(ang), sn = __sinf(ang);
    float x1 = src[h * DQK + DNc + 2 * i];
    float x2 = src[h * DQK + DNc + 2 * i + 1];
    size_t base = (((size_t)(b * Hc + h)) * Tc + t) * DQK + DNc;
    Qf[base + 2 * i]     = f2bf(x1 * cs - x2 * sn);
    Qf[base + 2 * i + 1] = f2bf(x1 * sn + x2 * cs);
  }
}

// ---------------------------------------------------------------------------
// k_rope: rope kv_raw[:,512:576]; write f32 cache cols 512..575 and broadcast
// bf16 into Kf[...,128:192] for all heads.  32 threads per (b,t) row.
// ---------------------------------------------------------------------------
__global__ void k_rope(const float* __restrict__ kv_raw,
                       float* __restrict__ cache,
                       unsigned short* __restrict__ Kf)
{
  const int row = blockIdx.x, b = row / Tc, t = row % Tc, i = threadIdx.x; // 0..31
  float freq = __expf(-__logf(10000.0f) * (float)(2 * i) / (float)DRc);
  float ang = (float)t * freq;
  float cs = __cosf(ang), sn = __sinf(ang);
  const float* src = kv_raw + (size_t)row * (KVLc + DRc) + KVLc;
  float x1 = src[2 * i], x2 = src[2 * i + 1];
  float r0 = x1 * cs - x2 * sn, r1 = x1 * sn + x2 * cs;
  cache[(size_t)row * (KVLc + DRc) + KVLc + 2 * i]     = r0;
  cache[(size_t)row * (KVLc + DRc) + KVLc + 2 * i + 1] = r1;
  unsigned short b0 = f2bf(r0), b1 = f2bf(r1);
  for (int h = 0; h < Hc; ++h) {
    size_t base = (((size_t)(b * Hc + h)) * Tc + t) * DQK + DNc;
    Kf[base + 2 * i] = b0;
    Kf[base + 2 * i + 1] = b1;
  }
}

// ---------------------------------------------------------------------------
// kvu repack: kvu (BT x H*256 f32) -> Kf nope bf16 [(b,h,t) x 192 cols 0..127]
// and V TRANSPOSED bf16 VfT[(b,h) x 128(dv) x T] so P*V B-frags are contiguous.
// ---------------------------------------------------------------------------
__global__ __launch_bounds__(256) void kvu_repack(
    const float* __restrict__ kvu,
    unsigned short* __restrict__ Kf, unsigned short* __restrict__ VfT)
{
  const int row = blockIdx.x, b = row / Tc, t = row % Tc, tid = threadIdx.x;
  const float* src = kvu + (size_t)row * (Hc * (DNc + DVc));
  for (int i = tid; i < Hc * (DNc + DVc); i += 256) {
    int h = i >> 8, c = i & 255;
    unsigned short v = f2bf(src[i]);
    size_t bh = (size_t)(b * Hc + h);
    if (c < DNc) Kf[(bh * Tc + t) * DQK + c] = v;
    else         VfT[(bh * 128 + (c - DNc)) * Tc + t] = v;
  }
}

// ---------------------------------------------------------------------------
// Flash attention (non-causal): block = 4 waves x 16 query rows = 64 queries,
// key tiles of 64.  S and O accumulated with WMMA bf16; online softmax.
// Qf/Kf: [(b,h,t) x 192] bf16;  VfT: [(b,h) x 128 x T] bf16;
// Obf:   [(b,t) x 2048] bf16  (== (B,T,H*DV) layout for the Wo GEMM)
// ---------------------------------------------------------------------------
__global__ __launch_bounds__(128) void mla_attn(
    const unsigned short* __restrict__ Qf,
    const unsigned short* __restrict__ Kf,
    const unsigned short* __restrict__ VfT,
    unsigned short* __restrict__ Obf)
{
  __shared__ unsigned short Kt[64 * 200];       // keys x 192, stride 200
  __shared__ unsigned short Vt[128 * 72];       // dv x 64 keys, stride 72
  __shared__ unsigned short Pst[4][16 * 72];    // per-wave P staging, stride 72

  const int bh = blockIdx.y;
  const int b = bh >> 4, h = bh & 15;
  const int q0 = blockIdx.x * 64;
  const int tid = threadIdx.x, wave = tid >> 5, lane = tid & 31;
  const int half = lane >> 4, l15 = lane & 15;
  const float scale = 0.0721687836487f;         // 1/sqrt(192)

  // Q fragments (6 along K=192) held in registers for the whole kernel
  Frag16 qf[6];
  {
    const unsigned short* qp = Qf + ((size_t)bh * Tc + (q0 + wave * 16 + l15)) * DQK;
#pragma unroll
    for (int kk = 0; kk < 6; ++kk) {
      const unsigned short* p = qp + kk * 32 + half * 8;
      qf[kk].q[0] = *(const uint4*)p;
      qf[kk].q[1] = *(const uint4*)(p + 16);
    }
  }

  const f32x8 fz = {0.f,0.f,0.f,0.f,0.f,0.f,0.f,0.f};
  f32x8 o[8];
  float m[8], l[8];
#pragma unroll
  for (int r = 0; r < 8; ++r) { o[r] = fz; m[r] = -3.0e38f; l[r] = 0.f; }

  const unsigned short* kg = Kf  + (size_t)bh * Tc * DQK;
  const unsigned short* vg = VfT + (size_t)bh * 128 * Tc;

  for (int kt = 0; kt < Tc; kt += 64) {
    // cooperative async tile loads (16B chunks straight into LDS)
    for (int i = tid; i < 64 * 24; i += 128) {           // K tile: 64 x 192
      int r = i / 24, c = (i % 24) * 8;
      g2l_b128(kg + (size_t)(kt + r) * DQK + c, &Kt[r * 200 + c]);
    }
    for (int i = tid; i < 128 * 8; i += 128) {           // V tile: 128(dv) x 64
      int r = i >> 3, c = (i & 7) * 8;
      g2l_b128(vg + (size_t)r * Tc + kt + c, &Vt[r * 72 + c]);
    }
    async_wait();
    __syncthreads();

    // S = Q * K^T  (16 queries x 64 keys per wave)
    f32x8 s[4] = {fz, fz, fz, fz};
#pragma unroll
    for (int kk = 0; kk < 6; ++kk)
#pragma unroll
      for (int nf = 0; nf < 4; ++nf) {
        Frag16 kfr;
        const unsigned short* p = &Kt[(nf * 16 + l15) * 200 + kk * 32 + half * 8];
        kfr.q[0] = *(const uint4*)p;
        kfr.q[1] = *(const uint4*)(p + 16);
        s[nf] = __builtin_amdgcn_wmma_f32_16x16x32_bf16(
            false, qf[kk].v, false, kfr.v, (short)0, s[nf], false, false);
      }

    // online softmax per row (row = r + half*8, col = lane&15 within frag)
#pragma unroll
    for (int r = 0; r < 8; ++r) {
      float s0 = s[0][r] * scale, s1 = s[1][r] * scale;
      float s2 = s[2][r] * scale, s3 = s[3][r] * scale;
      float rmax = fmaxf(fmaxf(s0, s1), fmaxf(s2, s3));
      rmax = fmaxf(rmax, __shfl_xor(rmax, 1, 32));
      rmax = fmaxf(rmax, __shfl_xor(rmax, 2, 32));
      rmax = fmaxf(rmax, __shfl_xor(rmax, 4, 32));
      rmax = fmaxf(rmax, __shfl_xor(rmax, 8, 32));
      float mn = fmaxf(m[r], rmax);
      float alpha = __expf(m[r] - mn);
      float p0 = __expf(s0 - mn), p1 = __expf(s1 - mn);
      float p2 = __expf(s2 - mn), p3 = __expf(s3 - mn);
      float rsum = p0 + p1 + p2 + p3;
      rsum += __shfl_xor(rsum, 1, 32);
      rsum += __shfl_xor(rsum, 2, 32);
      rsum += __shfl_xor(rsum, 4, 32);
      rsum += __shfl_xor(rsum, 8, 32);
      l[r] = l[r] * alpha + rsum;
      m[r] = mn;
#pragma unroll
      for (int ov = 0; ov < 8; ++ov) o[ov][r] *= alpha;
      unsigned short* pr = &Pst[wave][(r + half * 8) * 72];
      pr[ 0 + l15] = f2bf(p0);
      pr[16 + l15] = f2bf(p1);
      pr[32 + l15] = f2bf(p2);
      pr[48 + l15] = f2bf(p3);
    }
    __syncthreads();

    // O += P * V
#pragma unroll
    for (int kk = 0; kk < 2; ++kk) {
      Frag16 pf;
      const unsigned short* pp = &Pst[wave][l15 * 72 + kk * 32 + half * 8];
      pf.q[0] = *(const uint4*)pp;
      pf.q[1] = *(const uint4*)(pp + 16);
#pragma unroll
      for (int nv = 0; nv < 8; ++nv) {
        Frag16 vfr;
        const unsigned short* vp = &Vt[(nv * 16 + l15) * 72 + kk * 32 + half * 8];
        vfr.q[0] = *(const uint4*)vp;
        vfr.q[1] = *(const uint4*)(vp + 16);
        o[nv] = __builtin_amdgcn_wmma_f32_16x16x32_bf16(
            false, pf.v, false, vfr.v, (short)0, o[nv], false, false);
      }
    }
    __syncthreads();
  }

  // epilogue: O /= l, write bf16 in (B,T,H*DV) layout
#pragma unroll
  for (int r = 0; r < 8; ++r) {
    float inv = (l[r] > 0.f) ? (1.0f / l[r]) : 0.f;
    int trow = q0 + wave * 16 + r + half * 8;
    size_t base = ((size_t)b * Tc + trow) * (Hc * DVc) + (size_t)h * DVc;
#pragma unroll
    for (int nv = 0; nv < 8; ++nv)
      Obf[base + nv * 16 + l15] = f2bf(o[nv][r] * inv);
  }
}

// ---------------------------------------------------------------------------
// Host-side launcher
// ---------------------------------------------------------------------------
extern "C" void kernel_launch(void* const* d_in, const int* in_sizes, int n_in,
                              void* d_out, int out_size, void* d_ws, size_t ws_size,
                              hipStream_t stream) {
  (void)in_sizes; (void)n_in; (void)out_size; (void)ws_size;

  const float* x        = (const float*)d_in[0];
  const float* Wq_down  = (const float*)d_in[1];
  const float* q_ln_g   = (const float*)d_in[2];
  const float* q_ln_b   = (const float*)d_in[3];
  const float* Wq_up    = (const float*)d_in[4];
  const float* bq_up    = (const float*)d_in[5];
  const float* Wkv_down = (const float*)d_in[6];
  const float* kv_ln_g  = (const float*)d_in[7];
  const float* kv_ln_b  = (const float*)d_in[8];
  const float* Wkv_up   = (const float*)d_in[9];
  const float* bkv_up   = (const float*)d_in[10];
  const float* Wo       = (const float*)d_in[11];

  float* out   = (float*)d_out;                 // (B,T,C)
  float* cache = out + (size_t)BT * Cc;         // (B,T,KVL+DR)

  char* ws = (char*)d_ws;
  size_t off = 0;
  auto alloc = [&](size_t bytes) -> void* {
    void* p = ws + off;
    off = (off + bytes + 255) & ~(size_t)255;
    return p;
  };

  unsigned short* x_bf    = (unsigned short*)alloc((size_t)BT * Cc * 2);
  unsigned short* Wqd_bf  = (unsigned short*)alloc((size_t)QLc * Cc * 2);
  unsigned short* Wqu_bf  = (unsigned short*)alloc((size_t)Hc * DQK * QLc * 2);
  unsigned short* Wkvd_bf = (unsigned short*)alloc((size_t)(KVLc + DRc) * Cc * 2);
  unsigned short* Wkvu_bf = (unsigned short*)alloc((size_t)Hc * (DNc + DVc) * KVLc * 2);
  unsigned short* Wo_bf   = (unsigned short*)alloc((size_t)Cc * Hc * DVc * 2);
  unsigned short* cq_bf   = (unsigned short*)alloc((size_t)BT * QLc * 2);
  unsigned short* ckv_bf  = (unsigned short*)alloc((size_t)BT * KVLc * 2);
  unsigned short* Qf      = (unsigned short*)alloc((size_t)Bc * Hc * Tc * DQK * 2);
  unsigned short* Kfb     = (unsigned short*)alloc((size_t)Bc * Hc * Tc * DQK * 2);
  unsigned short* VfT     = (unsigned short*)alloc((size_t)Bc * Hc * 128 * Tc * 2);
  unsigned short* attn_bf = (unsigned short*)alloc((size_t)BT * Hc * DVc * 2);
  float* kv_raw = (float*)alloc((size_t)BT * (KVLc + DRc) * 4);
  // big f32 region reused serially: cq_raw (25MB) -> q_raw (50MB) -> kvu (67MB)
  float* big = (float*)alloc((size_t)BT * 4096 * 4);
  float* cq_raw = big;
  float* q_raw  = big;
  float* kvu    = big;

  // 1) convert inputs/weights to bf16
  cvt_f32_bf16<<<2048, 256, 0, stream>>>(x,        x_bf,    (size_t)BT * Cc);
  cvt_f32_bf16<<<2048, 256, 0, stream>>>(Wq_down,  Wqd_bf,  (size_t)QLc * Cc);
  cvt_f32_bf16<<<2048, 256, 0, stream>>>(Wq_up,    Wqu_bf,  (size_t)Hc * DQK * QLc);
  cvt_f32_bf16<<<2048, 256, 0, stream>>>(Wkv_down, Wkvd_bf, (size_t)(KVLc + DRc) * Cc);
  cvt_f32_bf16<<<2048, 256, 0, stream>>>(Wkv_up,   Wkvu_bf, (size_t)Hc * (DNc + DVc) * KVLc);
  cvt_f32_bf16<<<2048, 256, 0, stream>>>(Wo,       Wo_bf,   (size_t)Cc * Hc * DVc);

  // 2) q path: cq = LN(x @ Wq_down^T);  q = cq @ Wq_up^T + b;  rope+repack
  gemm_bf16_nt<<<dim3(QLc / 64, BT / 128), 256, 0, stream>>>(
      x_bf, Wqd_bf, nullptr, cq_raw, BT, QLc, Cc);
  ln_kernel<<<BT, 256, 0, stream>>>(cq_raw, QLc, QLc, q_ln_g, q_ln_b,
                                    cq_bf, QLc, nullptr, 0);
  gemm_bf16_nt<<<dim3(Hc * DQK / 64, BT / 128), 256, 0, stream>>>(
      cq_bf, Wqu_bf, bq_up, q_raw, BT, Hc * DQK, QLc);
  q_repack<<<BT, 256, 0, stream>>>(q_raw, Qf);

  // 3) kv path: kv = x @ Wkv_down^T; LN -> cache+ckv; rope -> cache+Kf
  gemm_bf16_nt<<<dim3((KVLc + DRc) / 64, BT / 128), 256, 0, stream>>>(
      x_bf, Wkvd_bf, nullptr, kv_raw, BT, KVLc + DRc, Cc);
  ln_kernel<<<BT, 256, 0, stream>>>(kv_raw, KVLc + DRc, KVLc, kv_ln_g, kv_ln_b,
                                    ckv_bf, KVLc, cache, KVLc + DRc);
  k_rope<<<BT, 32, 0, stream>>>(kv_raw, cache, Kfb);

  // 4) kvu = ckv @ Wkv_up^T + b;  split into K_nope / V^T
  gemm_bf16_nt<<<dim3(Hc * (DNc + DVc) / 64, BT / 128), 256, 0, stream>>>(
      ckv_bf, Wkvu_bf, bkv_up, kvu, BT, Hc * (DNc + DVc), KVLc);
  kvu_repack<<<BT, 256, 0, stream>>>(kvu, Kfb, VfT);

  // 5) attention
  mla_attn<<<dim3(Tc / 64, Bc * Hc), 128, 0, stream>>>(Qf, Kfb, VfT, attn_bf);

  // 6) output projection straight into d_out
  gemm_bf16_nt<<<dim3(Cc / 64, BT / 128), 256, 0, stream>>>(
      attn_bf, Wo_bf, nullptr, out, BT, Cc, Hc * DVc);
}
